// SelfAttention_72395968741829
// MI455X (gfx1250) — compile-verified, matrix operand order checked by reading, not run on hardware
//
#include <hip/hip_runtime.h>

// Problem constants (match the reference).
#define BB   2
#define TT   2048
#define DDIM 1024
#define HH   16
#define DKK  64
#define MMR  (BB * TT)   // 4096 rows of x / out

typedef __attribute__((ext_vector_type(16))) __bf16 bf16x16;
typedef __attribute__((ext_vector_type(2)))  __bf16 bf16x2;
typedef __attribute__((ext_vector_type(8)))  float  f32x8;

struct U8 { unsigned int u[8]; };   // raw storage for one 16x32 bf16 fragment

// Prefer the native packed converter (v_cvt_pk_bf16_f32) when the toolchain
// exposes it for gfx1250; fall back to the RNE bit-trick otherwise.
#if defined(__has_builtin)
#if __has_builtin(__builtin_amdgcn_cvt_pk_bf16_f32)
#define HAVE_CVT_PK_BF16 1
#endif
#endif

__device__ __forceinline__ unsigned short f2bf_bits(float f) {
  unsigned int u = __float_as_uint(f);
  unsigned int r = u + 0x7FFFu + ((u >> 16) & 1u);
  return (unsigned short)(r >> 16);
}

__device__ __forceinline__ unsigned int pack_bf(float lo, float hi) {
#ifdef HAVE_CVT_PK_BF16
  return __builtin_bit_cast(unsigned int, __builtin_amdgcn_cvt_pk_bf16_f32(lo, hi));
#else
  return (unsigned int)f2bf_bits(lo) | ((unsigned int)f2bf_bits(hi) << 16);
#endif
}

__device__ __forceinline__ unsigned short f2bf(float f) {
#ifdef HAVE_CVT_PK_BF16
  return (unsigned short)pack_bf(f, 0.f);
#else
  return f2bf_bits(f);
#endif
}

// Fragment dword i covers K = 8*half + {0,2,4,6} (i<4) or +16 (i>=4); pairs of
// consecutive K -> two 16-byte contiguous chunks per lane => 2x b128 loads.
// Identical index pattern for A- and B-fragments of 16-bit WMMA.
__device__ __forceinline__ U8 load_frag(const unsigned short* p, int half) {
  const uint4* q0 = (const uint4*)(p + 8 * half);
  const uint4* q1 = (const uint4*)(p + 8 * half + 16);
  uint4 a = *q0, b = *q1;
  U8 r;
  r.u[0] = a.x; r.u[1] = a.y; r.u[2] = a.z; r.u[3] = a.w;
  r.u[4] = b.x; r.u[5] = b.y; r.u[6] = b.z; r.u[7] = b.w;
  return r;
}

__device__ __forceinline__ f32x8 wmma_bf16(const U8& a, const U8& b, f32x8 c) {
  return __builtin_amdgcn_wmma_f32_16x16x32_bf16(
      false, __builtin_bit_cast(bf16x16, a),
      false, __builtin_bit_cast(bf16x16, b),
      (short)0, c, false, false);
}

// ---------------------------------------------------------------------------
// Bulk f32 -> bf16 conversion. n must be a multiple of 4.
// ---------------------------------------------------------------------------
__global__ void __launch_bounds__(256)
f32_to_bf16(const float* __restrict__ in, unsigned short* __restrict__ out, int n) {
  int i = (blockIdx.x * 256 + threadIdx.x) * 4;
  if (i >= n) return;
  float4 v = *(const float4*)(in + i);
  uint2 o;
  o.x = pack_bf(v.x, v.y);
  o.y = pack_bf(v.z, v.w);
  *(uint2*)(out + i) = o;
}

// ---------------------------------------------------------------------------
// GEMM: C[m,n] = sum_k A[m,k] * W[n,k] + bias[n]   (A,W bf16)
// Software-pipelined k-loop (ping-pong fragment sets).
// MODE 0: out f32 row-major [M,N]
// MODE 1: out bf16 head-major [B,H,T,DK]        (Q, K)
// MODE 2: out bf16 head-transposed [B,H,DK,T]   (V)
// block = 128 threads (4 waves); wave computes a 32(M) x 64(N) tile.
// ---------------------------------------------------------------------------
template <int MODE>
__global__ void __launch_bounds__(128)
gemm_bf16(const unsigned short* __restrict__ A, const unsigned short* __restrict__ W,
          const float* __restrict__ bias, void* __restrict__ out,
          int Ndim, int Kdim)
{
  const int lane = threadIdx.x & 31;
  const int wave = threadIdx.x >> 5;
  const int half = lane >> 4;
  const int l16  = lane & 15;
  const int nbase = blockIdx.x * 64;
  const int mbase = blockIdx.y * 128 + wave * 32;

  f32x8 acc[2][4] = {};

  const unsigned short* Ar0 = A + (size_t)(mbase + l16) * Kdim;
  const unsigned short* Ar1 = A + (size_t)(mbase + 16 + l16) * Kdim;
  const unsigned short* Wr[4];
#pragma unroll
  for (int nt = 0; nt < 4; ++nt)
    Wr[nt] = W + (size_t)(nbase + nt * 16 + l16) * Kdim;

  U8 a0[2], a1[2], b[2][4];

  auto loadset = [&](int s, int ks) {
    a0[s] = load_frag(Ar0 + ks, half);
    a1[s] = load_frag(Ar1 + ks, half);
#pragma unroll
    for (int nt = 0; nt < 4; ++nt) b[s][nt] = load_frag(Wr[nt] + ks, half);
  };
  auto mmaset = [&](int s) {
#pragma unroll
    for (int nt = 0; nt < 4; ++nt) {
      acc[0][nt] = wmma_bf16(a0[s], b[s][nt], acc[0][nt]);
      acc[1][nt] = wmma_bf16(a1[s], b[s][nt], acc[1][nt]);
    }
  };

  loadset(0, 0);
  for (int ks = 0; ks < Kdim - 64; ks += 64) {
    loadset(1, ks + 32);
    mmaset(0);
    loadset(0, ks + 64);
    mmaset(1);
  }
  loadset(1, Kdim - 32);
  mmaset(0);
  mmaset(1);

#pragma unroll
  for (int mt = 0; mt < 2; ++mt) {
#pragma unroll
    for (int nt = 0; nt < 4; ++nt) {
      int n = nbase + nt * 16 + l16;
      float bv = bias[n];
#pragma unroll
      for (int r = 0; r < 8; ++r) {
        int m = mbase + mt * 16 + r + 8 * half;
        float v = acc[mt][nt][r] + bv;
        if (MODE == 0) {
          ((float*)out)[(size_t)m * Ndim + n] = v;
        } else {
          int bb = m / TT, t = m % TT, h = n / DKK, dk = n % DKK;
          size_t idx = (MODE == 1)
              ? (((size_t)(bb * HH + h) * TT + t) * DKK + dk)
              : (((size_t)(bb * HH + h) * DKK + dk) * TT + t);
          ((unsigned short*)out)[idx] = f2bf(v);
        }
      }
    }
  }
}

// ---------------------------------------------------------------------------
// One 64-key attention tile, TRANSPOSED formulation:
//   S^T = K · Q^T   (A = K 16key x 32dk,  B = Q^T 32dk x 16q)
//   O^T = V^T · P^T (A = V^T 16dk x 32key, B = P^T 32key x 16q)
// S^T's D-layout puts each q-row's keys in ONE lane (split across halves), so
// softmax is a register max-tree + one shfl_xor(16); and P^T's B-fragment is
// built from the SAME lane's score vgprs -> no LDS, no cross-lane transpose.
// MASKED only for the single diagonal tile.
// ---------------------------------------------------------------------------
template <bool MASKED>
__device__ __forceinline__ void attn_tile(
    int k0, int qlane, int half, int l16,
    const unsigned short* __restrict__ kbh,
    const unsigned short* __restrict__ vbh,
    const U8 (&qf)[2],
    f32x8 (&O)[4], float& rmax, float& rsum)
{
  const float c = 0.125f * 1.44269504088896340736f;   // (1/sqrt(DK)) * log2(e)

  // ---- A-fragments of K (4 key sub-tiles x 2 dk-chunks), score WMMAs
  U8 kf[4][2];
#pragma unroll
  for (int nt = 0; nt < 4; ++nt) {
    const unsigned short* krow = kbh + (size_t)(k0 + nt * 16 + l16) * DKK;
#pragma unroll
    for (int t = 0; t < 2; ++t) kf[nt][t] = load_frag(krow + t * 32, half);
  }
  f32x8 st[4] = {};
#pragma unroll
  for (int nt = 0; nt < 4; ++nt)
#pragma unroll
    for (int t = 0; t < 2; ++t) st[nt] = wmma_bf16(kf[nt][t], qf[t], st[nt]);

  // ---- prefetch V^T A-fragments (independent of softmax; latency hidden)
  U8 vf[4][2];
#pragma unroll
  for (int nt = 0; nt < 4; ++nt) {
    const unsigned short* vrow = vbh + (size_t)(nt * 16 + l16) * TT + k0;
#pragma unroll
    for (int t = 0; t < 2; ++t) vf[nt][t] = load_frag(vrow + t * 32, half);
  }

  // ---- causal mask: lane's keys are k0 + nt*16 + r + 8*half, its q is qlane
  if (MASKED) {
#pragma unroll
    for (int nt = 0; nt < 4; ++nt)
#pragma unroll
      for (int r = 0; r < 8; ++r)
        st[nt][r] = (k0 + nt * 16 + r + 8 * half <= qlane)
                        ? st[nt][r] : -__builtin_inff();
  }

  // ---- online softmax: register reduction + one cross-half shuffle
  float mx = -__builtin_inff();
#pragma unroll
  for (int nt = 0; nt < 4; ++nt)
#pragma unroll
    for (int r = 0; r < 8; ++r) mx = fmaxf(mx, st[nt][r]);
  mx = fmaxf(mx, __shfl_xor(mx, 16, 32));
  float nm = fmaxf(rmax, mx);
  float fr = __builtin_exp2f((rmax - nm) * c);
  rmax = nm;

  float p[4][8];
  float ps = 0.f;
#pragma unroll
  for (int nt = 0; nt < 4; ++nt)
#pragma unroll
    for (int r = 0; r < 8; ++r) {
      p[nt][r] = __builtin_exp2f((st[nt][r] - nm) * c);
      ps += p[nt][r];
    }
  rsum = rsum * fr + ps;   // per-lane partial; merged across halves at the end
#pragma unroll
  for (int nt = 0; nt < 4; ++nt)
#pragma unroll
    for (int r = 0; r < 8; ++r) O[nt][r] = O[nt][r] * fr;

  // ---- P^T B-fragments: purely lane-local repack of score vgprs
  U8 pf[2];
#pragma unroll
  for (int t = 0; t < 2; ++t)
#pragma unroll
    for (int i = 0; i < 4; ++i) {
      pf[t].u[i]     = pack_bf(p[2 * t][2 * i],     p[2 * t][2 * i + 1]);
      pf[t].u[i + 4] = pack_bf(p[2 * t + 1][2 * i], p[2 * t + 1][2 * i + 1]);
    }

  // ---- O^T += V^T @ P^T
#pragma unroll
  for (int nt = 0; nt < 4; ++nt)
#pragma unroll
    for (int t = 0; t < 2; ++t) O[nt] = wmma_bf16(vf[nt][t], pf[t], O[nt]);
}

// ---------------------------------------------------------------------------
// Flash attention: per (b,h), one wave per 16-query tile, key tiles of 64.
// Q,K: bf16 [BH,T,DK]; V: bf16 [BH,DK,T]; out: bf16 [B,T,D] row-major.
// block = 128 threads (4 waves), grid = (T/64, B*H).  No LDS needed.
// ---------------------------------------------------------------------------
__global__ void __launch_bounds__(128)
flash_attn(const unsigned short* __restrict__ qb,
           const unsigned short* __restrict__ kb,
           const unsigned short* __restrict__ vb,
           unsigned short* __restrict__ attn)
{
  const int lane = threadIdx.x & 31;
  const int wave = threadIdx.x >> 5;
  const int half = lane >> 4;
  const int l16  = lane & 15;
  const int bh = blockIdx.y;
  const int b  = bh / HH, h = bh % HH;
  const int qbase = blockIdx.x * 64 + wave * 16;
  const int qlane = qbase + l16;           // this lane's query row

  const unsigned short* kbh = kb + (size_t)bh * TT * DKK;
  const unsigned short* vbh = vb + (size_t)bh * DKK * TT;

  // Q^T B-fragments (dk 0..31 and 32..63)
  const unsigned short* qrow = qb + ((size_t)bh * TT + qlane) * DKK;
  U8 qf[2];
#pragma unroll
  for (int t = 0; t < 2; ++t) qf[t] = load_frag(qrow + t * 32, half);

  f32x8 O[4] = {};
  float rmax = -__builtin_inff(), rsum = 0.f;

  // all tiles before klast are fully below the diagonal for this 16-row tile
  const int klast = ((qbase + 15) >> 6) << 6;
  for (int k0 = 0; k0 < klast; k0 += 64)
    attn_tile<false>(k0, qlane, half, l16, kbh, vbh, qf, O, rmax, rsum);
  attn_tile<true>(klast, qlane, half, l16, kbh, vbh, qf, O, rmax, rsum);

  // ---- merge halves' row sums, normalize, write bf16 [B,T,D] as b32 pairs
  float tot = rsum + __shfl_xor(rsum, 16, 32);
  float inv = 1.0f / tot;
  unsigned short* orow = attn + ((size_t)b * TT + qlane) * DDIM + h * DKK + 8 * half;
#pragma unroll
  for (int nt = 0; nt < 4; ++nt)
#pragma unroll
    for (int rp = 0; rp < 4; ++rp) {
      unsigned int w = pack_bf(O[nt][2 * rp] * inv, O[nt][2 * rp + 1] * inv);
      *(unsigned int*)(orow + nt * 16 + 2 * rp) = w;
    }
}

// ---------------------------------------------------------------------------
extern "C" void kernel_launch(void* const* d_in, const int* in_sizes, int n_in,
                              void* d_out, int out_size, void* d_ws, size_t ws_size,
                              hipStream_t stream) {
  (void)in_sizes; (void)n_in; (void)out_size; (void)ws_size;
  const float* x  = (const float*)d_in[0];
  // d_in[1] = causal mask, applied analytically
  const float* Wq = (const float*)d_in[2];
  const float* bq = (const float*)d_in[3];
  const float* Wk = (const float*)d_in[4];
  const float* bk = (const float*)d_in[5];
  const float* Wv = (const float*)d_in[6];
  const float* bv = (const float*)d_in[7];
  const float* Wo = (const float*)d_in[8];
  const float* bo = (const float*)d_in[9];

  const size_t NX = (size_t)MMR * DDIM;    // 4M elements
  const size_t NW = (size_t)DDIM * DDIM;   // 1M elements

  unsigned short* xbf  = (unsigned short*)d_ws;       // 8 MB
  unsigned short* wqb  = xbf + NX;                    // 2 MB
  unsigned short* wkb  = wqb + NW;                    // 2 MB
  unsigned short* wvb  = wkb + NW;                    // 2 MB
  unsigned short* wob  = wvb + NW;                    // 2 MB
  unsigned short* qbuf = wob + NW;                    // 8 MB bf16 [BH,T,DK]
  unsigned short* kbuf = qbuf + NX;                   // 8 MB bf16 [BH,T,DK]
  unsigned short* vbuf = kbuf + NX;                   // 8 MB bf16 [BH,DK,T]
  unsigned short* abuf = vbuf + NX;                   // 8 MB bf16 [B,T,D]

  // one-time f32 -> bf16 conversions (bandwidth-trivial)
  f32_to_bf16<<<(int)(NX / 4 + 255) / 256, 256, 0, stream>>>(x,  xbf, (int)NX);
  f32_to_bf16<<<(int)(NW / 4 + 255) / 256, 256, 0, stream>>>(Wq, wqb, (int)NW);
  f32_to_bf16<<<(int)(NW / 4 + 255) / 256, 256, 0, stream>>>(Wk, wkb, (int)NW);
  f32_to_bf16<<<(int)(NW / 4 + 255) / 256, 256, 0, stream>>>(Wv, wvb, (int)NW);
  f32_to_bf16<<<(int)(NW / 4 + 255) / 256, 256, 0, stream>>>(Wo, wob, (int)NW);

  dim3 gg(DDIM / 64, MMR / 128);   // 16 x 32 blocks, 128 threads each
  gemm_bf16<1><<<gg, 128, 0, stream>>>(xbf, wqb, bq, qbuf, DDIM, DDIM);
  gemm_bf16<1><<<gg, 128, 0, stream>>>(xbf, wkb, bk, kbuf, DDIM, DDIM);
  gemm_bf16<2><<<gg, 128, 0, stream>>>(xbf, wvb, bv, vbuf, DDIM, DDIM);

  flash_attn<<<dim3(TT / 64, BB * HH), 128, 0, stream>>>(qbuf, kbuf, vbuf, abuf);

  gemm_bf16<0><<<gg, 128, 0, stream>>>(abuf, wob, bo, d_out, DDIM, DDIM);
}
